// MyVMLMFgCellg2_67027259621391
// MI455X (gfx1250) — matrix operand order repaired and checked by reading, not compiled, not run on hardware
//
#include <hip/hip_runtime.h>
#include <hip/hip_bf16.h>

// CDNA5 / gfx1250 WMMA types (wave32)
typedef __attribute__((ext_vector_type(16))) __bf16 v16bf;
typedef __attribute__((ext_vector_type(8)))  float  v8f;

#define B_ROWS 8192
#define IN_DIM 1024
#define H_DIM  1024
#define HG     512
#define T_COLS 320   // 64 (x-path rank) + 4*64 (h-path: i in {0,1} x g in {0,1})

__device__ __forceinline__ float sigmoidf_(float x) { return 1.0f / (1.0f + __expf(-x)); }

// Load a 16x32 bf16 A-fragment from an f32 row-major matrix.
// Layout (16-bit A 16x32): lane&15 = M; lanes 0-15 cover K=0..15,
// lanes 16-31 cover K=16..31 (matches B swizzle). Per lane: one contiguous 16-float run.
__device__ __forceinline__ v16bf load_a_frag(const float* __restrict__ rowp, int c0, int half) {
  const float* p0 = rowp + c0 + half * 16;
  const float* p1 = p0 + 8;
  v16bf r;
#pragma unroll
  for (int e = 0; e < 8; ++e) r[e] = (__bf16)p0[e];
#pragma unroll
  for (int e = 0; e < 8; ++e) r[8 + e] = (__bf16)p1[e];
  return r;
}

// ---------------------------------------------------------------------------
// Stage 1: T[b, 0:64]   = x @ u_x
//          T[b, 64+..]  = h @ Uh(i,g)   (rolled-group low-rank, zero-padded K)
// Block = 256 thr = 8 waves, M-tile 128, N = 320 (20 WMMA tiles/wave), K = 1024
// in 8 chunks of 128; weights converted+swizzled into LDS per chunk.
// ---------------------------------------------------------------------------
__global__ __launch_bounds__(256) void k1_lowrank_stage1(
    const float* __restrict__ x, const float* __restrict__ h,
    const float* __restrict__ u_x, const float* __restrict__ u_h_0,
    const float* __restrict__ u_h_1, float* __restrict__ T)
{
  extern __shared__ __align__(64) char smem_raw[];
  __bf16* sW = (__bf16*)smem_raw;             // [4 ksteps][20 ntiles][32 lanes][16] = 40960 bf16 (80 KB)

  const int tid  = threadIdx.x;
  const int lane = tid & 31;
  const int wv   = tid >> 5;                  // 0..7
  const int half = lane >> 4;
  const int mrow = blockIdx.x * 128 + wv * 16 + (lane & 15);
  const float* xrow = x + (size_t)mrow * IN_DIM;
  const float* hrow = h + (size_t)mrow * IN_DIM;

  v8f zero;
#pragma unroll
  for (int e = 0; e < 8; ++e) zero[e] = 0.0f;
  v8f acc[20];
#pragma unroll
  for (int t = 0; t < 20; ++t) acc[t] = zero;

  for (int chunk = 0; chunk < 8; ++chunk) {
    if (chunk < 7) {  // hint next K-chunk of activations into cache
      __builtin_prefetch(xrow + (chunk + 1) * 128, 0, 0);
      __builtin_prefetch(hrow + (chunk + 1) * 128, 0, 0);
    }
    __syncthreads();
    // Cooperative fp32->bf16 convert + fragment-order swizzle of weight chunk.
    for (int idx = tid; idx < 40960; idx += 256) {
      int e    = idx & 15;
      int ln   = (idx >> 4) & 31;
      int rest = idx >> 9;
      int t    = rest % 20;
      int s    = rest / 20;
      int kpat = (ln >> 4) * 16 + e;          // B-frag: lane half selects 16-K half
      int k    = chunk * 128 + s * 32 + kpat; // global K row
      int n    = t * 16 + (ln & 15);          // global N col (0..319)
      float w;
      if (n < 64) {
        w = u_x[k * 64 + n];
      } else {
        int cc = n - 64, i = cc >> 7, g = (cc >> 6) & 1, r = cc & 63;
        int src_g = (g + i) & 1;              // rolled group feeding this block
        if ((k >> 9) == src_g) {
          const float* u = i ? u_h_1 : u_h_0;
          w = u[(g * HG + (k & 511)) * 64 + r];
        } else {
          w = 0.0f;                           // zero-pad inactive group half
        }
      }
      sW[idx] = (__bf16)w;
    }
    __syncthreads();

#pragma unroll
    for (int s = 0; s < 4; ++s) {
      int kc = chunk * 128 + s * 32;
      v16bf ax = load_a_frag(xrow, kc, half);
      v16bf ah = load_a_frag(hrow, kc, half);
#pragma unroll
      for (int t = 0; t < 20; ++t) {
        v16bf bw = *(const v16bf*)(sW + ((s * 20 + t) * 32 + lane) * 16);
        acc[t] = __builtin_amdgcn_wmma_f32_16x16x32_bf16(
            false, (t < 4) ? ax : ah, false, bw, (short)0, acc[t], false, false);
      }
    }
  }

  // C/D layout: VGPR r, lanes 0-15 -> M=r, N=lane; lanes 16-31 -> M=8+r.
  const int mbase = blockIdx.x * 128 + wv * 16;
#pragma unroll
  for (int t = 0; t < 20; ++t) {
    int n = t * 16 + (lane & 15);
#pragma unroll
    for (int r = 0; r < 8; ++r) {
      int m = mbase + r + 8 * half;
      T[(size_t)m * T_COLS + n] = acc[t][r];
    }
  }
}

// ---------------------------------------------------------------------------
// Stage 2: per block: one (g, 64-wide d-slab), all 4 gates (N = 4q x 4 x 16),
// K = 192 active rows of the combined V factor (no zero padding). Fused gate
// epilogue: sigmoid/tanh, c update, h_next/c_next stores — all in registers.
// ---------------------------------------------------------------------------
__global__ __launch_bounds__(256) void k2_lowrank_stage2(
    const float* __restrict__ T, const float* __restrict__ v_x,
    const float* __restrict__ v_h_0, const float* __restrict__ v_h_1,
    const float* __restrict__ bias_x, const float* __restrict__ bias_h,
    const float* __restrict__ c, float* __restrict__ out_h, float* __restrict__ out_c)
{
  extern __shared__ __align__(64) char smem_raw[];
  __bf16* sV = (__bf16*)smem_raw;             // [6 ksteps][16 ntiles][32 lanes][16] = 49152 bf16 (96 KB)

  const int tid   = threadIdx.x;
  const int lane  = tid & 31;
  const int wv    = tid >> 5;
  const int half  = lane >> 4;
  const int g     = blockIdx.y & 1;
  const int dbase = (blockIdx.y >> 1) * 64;

  // Cooperative convert+swizzle of the K=192 x N=256 second-stage factor.
  for (int idx = tid; idx < 49152; idx += 256) {
    int e    = idx & 15;
    int ln   = (idx >> 4) & 31;
    int t    = (idx >> 9) & 15;
    int s    = idx >> 13;
    int kpat = (ln >> 4) * 16 + e;
    int klog = s * 32 + kpat;                 // 0..191 logical K
    int q    = t >> 2;                        // gate index
    int d    = dbase + (t & 3) * 16 + (ln & 15);
    float w;
    if (klog < 64) {
      int j = q * 1024 + g * 512 + d;         // gx = t_x @ v_x^T
      w = v_x[j * 64 + klog];
    } else if (klog < 128) {
      int r = klog - 64;
      w = v_h_0[(g * 64 + r) * 2048 + q * 512 + d];
    } else {
      int r = klog - 128;
      w = v_h_1[(g * 64 + r) * 2048 + q * 512 + d];
    }
    sV[idx] = (__bf16)w;
  }
  __syncthreads();

  const int mrow = blockIdx.x * 128 + wv * 16 + (lane & 15);
  const float* Trow = T + (size_t)mrow * T_COLS;

  v8f zero;
#pragma unroll
  for (int e = 0; e < 8; ++e) zero[e] = 0.0f;
  v8f acc[16];
#pragma unroll
  for (int t = 0; t < 16; ++t) acc[t] = zero;

#pragma unroll
  for (int s = 0; s < 6; ++s) {
    // T-column bases of the three active 64-row blocks (t_x, t_h0[g], t_h1[g])
    const int cb = (s < 2) ? s * 32
                 : (s < 4) ? 64 + g * 64 + (s - 2) * 32
                           : 192 + g * 64 + (s - 4) * 32;
    v16bf a = load_a_frag(Trow, cb, half);
#pragma unroll
    for (int t = 0; t < 16; ++t) {
      v16bf bw = *(const v16bf*)(sV + ((s * 16 + t) * 32 + lane) * 16);
      acc[t] = __builtin_amdgcn_wmma_f32_16x16x32_bf16(
          false, a, false, bw, (short)0, acc[t], false, false);
    }
  }

  // Fused LSTM gate epilogue (f,i,n,o live in acc[dblk], acc[4+..], ...).
  const int mbase = blockIdx.x * 128 + wv * 16;
#pragma unroll
  for (int dblk = 0; dblk < 4; ++dblk) {
    int hcol = g * 512 + dbase + dblk * 16 + (lane & 15);
    float bf = bias_x[hcol]        + bias_h[hcol];
    float bi = bias_x[1024 + hcol] + bias_h[1024 + hcol];
    float bn = bias_x[2048 + hcol] + bias_h[2048 + hcol];
    float bo = bias_x[3072 + hcol] + bias_h[3072 + hcol];
#pragma unroll
    for (int r = 0; r < 8; ++r) {
      int m = mbase + r + 8 * half;
      size_t off = (size_t)m * H_DIM + hcol;
      float fg = sigmoidf_(acc[dblk][r]      + bf);
      float ig = sigmoidf_(acc[4 + dblk][r]  + bi);
      float ng = tanhf    (acc[8 + dblk][r]  + bn);
      float og = sigmoidf_(acc[12 + dblk][r] + bo);
      float cn = fg * c[off] + ig * ng;
      out_h[off] = og * tanhf(cn);
      out_c[off] = cn;
    }
  }
}

extern "C" void kernel_launch(void* const* d_in, const int* in_sizes, int n_in,
                              void* d_out, int out_size, void* d_ws, size_t ws_size,
                              hipStream_t stream) {
  const float* x      = (const float*)d_in[0];
  const float* h      = (const float*)d_in[1];
  const float* c      = (const float*)d_in[2];
  const float* u_x    = (const float*)d_in[3];
  const float* v_x    = (const float*)d_in[4];
  const float* u_h_0  = (const float*)d_in[5];
  const float* v_h_0  = (const float*)d_in[6];
  const float* u_h_1  = (const float*)d_in[7];
  const float* v_h_1  = (const float*)d_in[8];
  const float* bias_x = (const float*)d_in[9];
  const float* bias_h = (const float*)d_in[10];

  float* out_h = (float*)d_out;
  float* out_c = out_h + (size_t)B_ROWS * H_DIM;
  float* T     = (float*)d_ws;   // 8192 x 320 f32 = 10.5 MB (stays in 192 MB L2)

  k1_lowrank_stage1<<<dim3(B_ROWS / 128), dim3(256), 40960 * sizeof(__bf16), stream>>>(
      x, h, u_x, u_h_0, u_h_1, T);
  k2_lowrank_stage2<<<dim3(B_ROWS / 128, 16), dim3(256), 49152 * sizeof(__bf16), stream>>>(
      T, v_x, v_h_0, v_h_1, bias_x, bias_h, c, out_h, out_c);
}
